// SimilarityWeightedAggregation_19490561589455
// MI455X (gfx1250) — matrix operand order repaired and checked by reading, not compiled, not run on hardware
//
#include <hip/hip_runtime.h>
#include <stdint.h>

typedef __attribute__((ext_vector_type(16))) __bf16 v16bf;
typedef __attribute__((ext_vector_type(8)))  float  v8f;

union Frag { uint4 u[2]; v16bf v; };

__device__ __forceinline__ unsigned short f2bf(float f) {
    unsigned int u = __float_as_uint(f);
    unsigned int r = u + 0x7FFFu + ((u >> 16) & 1u);   // RNE
    return (unsigned short)(r >> 16);
}
__device__ __forceinline__ float bf2f(unsigned short h) {
    return __uint_as_float(((unsigned int)h) << 16);
}

#define NPIX 16384
#define C    1024

// ---------------- K0: init hmax ----------------
__global__ void k_init_hmax(int* hmax) {
    if (threadIdx.x < 16) hmax[threadIdx.x] = 0;   // exp() > 0, so 0 is a safe identity
}

// ---------------- K1: channel-wise global max pool of keys & values ----------------
__global__ void k_maxpool(const float* __restrict__ keys, const float* __restrict__ values,
                          float* __restrict__ kpmax, float* __restrict__ vpmax) {
    __shared__ float red[256];
    int t = threadIdx.x;
    int c = blockIdx.x & 1023;
    int which = blockIdx.x >> 10;
    const float* src = which ? values : keys;
    float m = -3.4e38f;
    for (int p = t; p < NPIX; p += 256) m = fmaxf(m, src[(size_t)c * NPIX + p]);
    red[t] = m; __syncthreads();
    for (int s = 128; s > 0; s >>= 1) {
        if (t < s) red[t] = fmaxf(red[t], red[t + s]);
        __syncthreads();
    }
    if (t == 0) (which ? vpmax : kpmax)[c] = red[0];
}

// ---------------- K2: kproj/vproj = w_in @ pooled + b_in ----------------
__global__ void k_proj_kv(const float* __restrict__ w_in, const float* __restrict__ b_in,
                          const float* __restrict__ kpmax, const float* __restrict__ vpmax,
                          float* __restrict__ kproj, float* __restrict__ vproj) {
    __shared__ float kp_s[C], vp_s[C];
    int t = threadIdx.x;
    for (int i = t; i < C; i += 256) { kp_s[i] = kpmax[i]; vp_s[i] = vpmax[i]; }
    __syncthreads();
    int o = blockIdx.x * 256 + t;
    float ak = 0.f, av = 0.f;
    const float4* wrow = (const float4*)(w_in + (size_t)o * C);
    for (int c4 = 0; c4 < C / 4; ++c4) {
        float4 w = wrow[c4];
        int c = c4 * 4;
        ak += w.x * kp_s[c] + w.y * kp_s[c + 1] + w.z * kp_s[c + 2] + w.w * kp_s[c + 3];
        av += w.x * vp_s[c] + w.y * vp_s[c + 1] + w.z * vp_s[c + 2] + w.w * vp_s[c + 3];
    }
    kproj[o] = ak + b_in[o];
    vproj[o] = av + b_in[o];
}

// ---------------- K3: LayerNorm of kproj -> gk[o]=g*kf, per-head G/B constants ----------------
__global__ void k_ln_k(const float* __restrict__ kproj, const float* __restrict__ ln_g,
                       const float* __restrict__ ln_b, float* __restrict__ gk,
                       float* __restrict__ GB) {
    __shared__ float sA[C], sB[C];
    int t = threadIdx.x;
    float v = kproj[t];
    sA[t] = v; sB[t] = v * v; __syncthreads();
    for (int s = 512; s > 0; s >>= 1) {
        if (t < s) { sA[t] += sA[t + s]; sB[t] += sB[t + s]; }
        __syncthreads();
    }
    float mu = sA[0] * (1.f / 1024.f);
    float var = sB[0] * (1.f / 1024.f) - mu * mu;
    float rs = rsqrtf(var + 1e-5f);
    __syncthreads();
    float g = ln_g[t], b = ln_b[t];
    float kf = (v - mu) * rs * g + b;
    sA[t] = g * kf; sB[t] = b * kf;
    gk[t] = g * kf;
    __syncthreads();
    if (t < 16) {
        float G = 0.f, B = 0.f;
        for (int j = 0; j < 64; ++j) { G += sA[t * 64 + j]; B += sB[t * 64 + j]; }
        GB[t] = G; GB[16 + t] = B;
    }
}

// ---------------- K4: wv[o][h] = sum_d w_out[o][h*64+d] * vproj[h*64+d] ----------------
__global__ void k_wv(const float* __restrict__ w_out, const float* __restrict__ vproj,
                     float* __restrict__ wv) {
    __shared__ float vp_s[C];
    int t = threadIdx.x;
    for (int i = t; i < C; i += 256) vp_s[i] = vproj[i];
    __syncthreads();
    int o = blockIdx.x * 256 + t;
    const float4* wrow = (const float4*)(w_out + (size_t)o * C);
    for (int h = 0; h < 16; ++h) {
        float acc = 0.f;
        for (int d4 = 0; d4 < 16; ++d4) {
            float4 w = wrow[h * 16 + d4];
            int d = h * 64 + d4 * 4;
            acc += w.x * vp_s[d] + w.y * vp_s[d + 1] + w.z * vp_s[d + 2] + w.w * vp_s[d + 3];
        }
        wv[o * 16 + h] = acc;
    }
}

// ---------------- K4b: convert w_in (f32 row-major [o][c]) -> bf16 ----------------
__global__ void k_cvt_win(const float* __restrict__ w_in, unsigned short* __restrict__ winb) {
    size_t i = ((size_t)blockIdx.x * 256 + threadIdx.x) * 4;
    float4 f = *(const float4*)(w_in + i);
    uint2 o;
    o.x = (unsigned int)f2bf(f.x) | ((unsigned int)f2bf(f.y) << 16);
    o.y = (unsigned int)f2bf(f.z) | ((unsigned int)f2bf(f.w) << 16);
    *(uint2*)(winb + i) = o;
}

// ---------------- K4c: transpose query [c][p] f32 -> qt [p][c] bf16 ----------------
__global__ void k_transpose_q(const float* __restrict__ query, unsigned short* __restrict__ qt) {
    __shared__ float tile[32][33];
    int t = threadIdx.x;
    int pbase = blockIdx.x * 32, cbase = blockIdx.y * 32;
    int lo = t & 31, hi = t >> 5;   // 8 rows per pass
    for (int r = 0; r < 4; ++r) {
        int c = hi + r * 8;
        tile[c][lo] = query[(size_t)(cbase + c) * NPIX + pbase + lo];
    }
    __syncthreads();
    for (int r = 0; r < 4; ++r) {
        int p = hi + r * 8;
        qt[(size_t)(pbase + p) * C + cbase + lo] = f2bf(tile[lo][p]);
    }
}

// ---------------- K5: bf16 WMMA GEMM  qproj[o][p] = w_in @ query  (bias deferred) ----------------
// block tile: M=256, N=128, K-step=32; 8 waves, wave tile 128x32 (8x2 subtiles).
// Double-buffered async global->LDS DMA (ASYNCcnt): DMA for chunk i+1 overlaps
// the 16 WMMAs/wave of chunk i.
#define LDSW 40   // ushorts per LDS row (80B, padded from 64B)
__global__ void __launch_bounds__(256)
k_gemm_qproj(const unsigned short* __restrict__ Wb, const unsigned short* __restrict__ Qt,
             unsigned short* __restrict__ qproj) {
    __shared__ unsigned short As[2][256 * LDSW];
    __shared__ unsigned short Bs[2][128 * LDSW];
    int t = threadIdx.x;
    int pbase = blockIdx.x * 128;
    int mbase = blockIdx.y * 256;
    int wid = t >> 5, lane = t & 31;
    int wm = wid >> 2, wn = wid & 3;
    int g = lane >> 4, ln16 = lane & 15;

    // per-thread LDS byte offsets (flat->LDS keeps the offset in addr[31:0])
    int brow = t >> 1, bhalf = t & 1;
    unsigned ldsA0 = (unsigned)(size_t)(void*)&As[0][t * LDSW];
    unsigned ldsA1 = (unsigned)(size_t)(void*)&As[1][t * LDSW];
    unsigned ldsB0 = (unsigned)(size_t)(void*)&Bs[0][brow * LDSW + bhalf * 16];
    unsigned ldsB1 = (unsigned)(size_t)(void*)&Bs[1][brow * LDSW + bhalf * 16];
    const unsigned short* gA = Wb + (size_t)(mbase + t) * C;           // + kc
    const unsigned short* gB = Qt + (size_t)(pbase + brow) * C + bhalf * 16;  // + kc

#define ISSUE_DMA(la, lb, kc_)                                              \
    do {                                                                    \
        unsigned long long ga = (unsigned long long)(gA + (kc_));           \
        asm volatile(                                                       \
            "global_load_async_to_lds_b128 %0, %1, off\n\t"                 \
            "global_load_async_to_lds_b128 %0, %1, off offset:16\n\t"       \
            "global_load_async_to_lds_b128 %0, %1, off offset:32\n\t"       \
            "global_load_async_to_lds_b128 %0, %1, off offset:48"           \
            :: "v"(la), "v"(ga) : "memory");                                \
        unsigned long long gb = (unsigned long long)(gB + (kc_));           \
        asm volatile(                                                       \
            "global_load_async_to_lds_b128 %0, %1, off\n\t"                 \
            "global_load_async_to_lds_b128 %0, %1, off offset:16"           \
            :: "v"(lb), "v"(gb) : "memory");                                \
    } while (0)

    v8f acc[8][2];
#pragma unroll
    for (int mi = 0; mi < 8; ++mi)
#pragma unroll
        for (int ni = 0; ni < 2; ++ni) acc[mi][ni] = (v8f){};

    // prologue: fill buffer 0 with the first K-chunk
    ISSUE_DMA(ldsA0, ldsB0, 0);
    asm volatile("s_wait_asynccnt 0" ::: "memory");
    __syncthreads();

    for (int kc = 0; kc < 1024; kc += 32) {
        int b = (kc >> 5) & 1;
        if (kc + 32 < 1024) {   // DMA next chunk into the other buffer (overlaps compute)
            ISSUE_DMA(b ? ldsA0 : ldsA1, b ? ldsB0 : ldsB1, kc + 32);
        }
        if (kc + 64 < 1024) {   // pull the chunk after that toward L2 (global_prefetch_b8)
            __builtin_prefetch(gA + kc + 64, 0, 1);
            __builtin_prefetch(Qt + (size_t)(pbase + (t & 127)) * C + kc + 64, 0, 1);
        }

        const unsigned short* Ab = As[b];
        const unsigned short* Bb = Bs[b];
        Frag bf0, bf1;
        {   // B fragment: lane n=ln16, halves hold K = g*16 .. g*16+15 (contiguous)
            const uint4* pb = (const uint4*)(&Bb[(wn * 32 + ln16) * LDSW + g * 16]);
            bf0.u[0] = pb[0]; bf0.u[1] = pb[1];
            pb = (const uint4*)(&Bb[(wn * 32 + 16 + ln16) * LDSW + g * 16]);
            bf1.u[0] = pb[0]; bf1.u[1] = pb[1];
        }
#pragma unroll
        for (int mi = 0; mi < 8; ++mi) {
            Frag af;   // A fragment: K = g*16 .. g*16+15 (contiguous)
            int row = wm * 128 + mi * 16 + ln16;
            af.u[0] = *(const uint4*)(&Ab[row * LDSW + g * 16]);
            af.u[1] = *(const uint4*)(&Ab[row * LDSW + g * 16 + 8]);
            acc[mi][0] = __builtin_amdgcn_wmma_f32_16x16x32_bf16(
                false, af.v, false, bf0.v, (short)0, acc[mi][0], false, false);
            acc[mi][1] = __builtin_amdgcn_wmma_f32_16x16x32_bf16(
                false, af.v, false, bf1.v, (short)0, acc[mi][1], false, false);
        }
        // next buffer must be fully DMA'd, and everyone must be done reading
        // this buffer, before the following iteration
        asm volatile("s_wait_asynccnt 0" ::: "memory");
        __syncthreads();
    }
#undef ISSUE_DMA

    // store bf16 (C/D layout: VGPR r -> M = r + 8*g, N = ln16)
#pragma unroll
    for (int mi = 0; mi < 8; ++mi)
#pragma unroll
        for (int ni = 0; ni < 2; ++ni)
#pragma unroll
            for (int r = 0; r < 8; ++r) {
                int o = mbase + wm * 128 + mi * 16 + g * 8 + r;
                int p = pbase + wn * 32 + ni * 16 + ln16;
                qproj[(size_t)o * NPIX + p] = f2bf(acc[mi][ni][r]);
            }
}

// ---------------- K6: per-pixel LN + head logits + exp + per-head max ----------------
__global__ void __launch_bounds__(256)
k_attn(const unsigned short* __restrict__ qproj, const float* __restrict__ b_in,
       const float* __restrict__ gk, const float* __restrict__ GB,
       int* __restrict__ hmax_g, float* __restrict__ attn) {
    __shared__ float gk_s[C], bin_s[C];
    __shared__ float red[16 * 32 * 8];     // [h][px][og]
    __shared__ float r1[256], r2[256];     // [px][og]
    __shared__ float mu_s[32], rs_s[32];
    __shared__ float G_s[16], B_s[16];
    __shared__ int   hmax_s[16];

    int t = threadIdx.x;
    int og = t >> 5, px = t & 31;
    int pbase = blockIdx.x * 32;
    int p = pbase + px;

    for (int i = t; i < C; i += 256) { gk_s[i] = gk[i]; bin_s[i] = b_in[i]; }
    if (t < 16) { G_s[t] = GB[t]; B_s[t] = GB[16 + t]; hmax_s[t] = 0; }
    __syncthreads();

    float hacc[16];
    float s1 = 0.f, s2 = 0.f;
#pragma unroll
    for (int h = 0; h < 16; ++h) {
        float a = 0.f;
#pragma unroll
        for (int j = 0; j < 8; ++j) {
            int o = h * 64 + j * 8 + og;
            float v = bf2f(qproj[(size_t)o * NPIX + p]) + bin_s[o];
            s1 += v; s2 += v * v;
            a += v * gk_s[o];
        }
        hacc[h] = a;
    }
    r1[px * 8 + og] = s1; r2[px * 8 + og] = s2;
#pragma unroll
    for (int h = 0; h < 16; ++h) red[(h * 32 + px) * 8 + og] = hacc[h];
    __syncthreads();

    if (t < 32) {
        float a = 0.f, b = 0.f;
        for (int j = 0; j < 8; ++j) { a += r1[t * 8 + j]; b += r2[t * 8 + j]; }
        float mu = a * (1.f / 1024.f);
        float var = b * (1.f / 1024.f) - mu * mu;
        mu_s[t] = mu; rs_s[t] = rsqrtf(var + 1e-5f);
    }
    __syncthreads();

    for (int idx = t; idx < 512; idx += 256) {
        int h = idx >> 5, px2 = idx & 31;
        float S = 0.f;
        for (int j = 0; j < 8; ++j) S += red[(h * 32 + px2) * 8 + j];
        float mu = mu_s[px2], rs = rs_s[px2];
        float logit = rs * S - rs * mu * G_s[h] + B_s[h];
        float a = __expf(logit * 0.03125f);          // 1024^-0.5
        attn[(size_t)h * NPIX + pbase + px2] = a;
        atomicMax(&hmax_s[h], __float_as_int(a));    // a > 0: int compare == float compare
    }
    __syncthreads();
    if (t < 16) atomicMax(&hmax_g[t], hmax_s[t]);
}

// ---------------- K7: out[o][p] = b_out[o] + sum_h wv[o][h] * attn[h][p] / hmax[h] ----------------
__global__ void k_out(const float* __restrict__ wv, const int* __restrict__ hmax,
                      const float* __restrict__ attn, const float* __restrict__ b_out,
                      float* __restrict__ out) {
    __shared__ float wl[16];
    int t = threadIdx.x;
    int o = blockIdx.y;
    int p = blockIdx.x * 256 + t;
    if (t < 16) wl[t] = wv[o * 16 + t] / __int_as_float(hmax[t]);
    __syncthreads();
    float sum = b_out[o];
#pragma unroll
    for (int h = 0; h < 16; ++h) sum += wl[h] * attn[(size_t)h * NPIX + p];
    out[(size_t)o * NPIX + p] = sum;
}

// ---------------- host launcher ----------------
extern "C" void kernel_launch(void* const* d_in, const int* in_sizes, int n_in,
                              void* d_out, int out_size, void* d_ws, size_t ws_size,
                              hipStream_t stream) {
    const float* query  = (const float*)d_in[0];
    const float* keys   = (const float*)d_in[1];
    const float* values = (const float*)d_in[2];
    const float* w_in   = (const float*)d_in[3];
    const float* b_in   = (const float*)d_in[4];
    const float* w_out  = (const float*)d_in[5];
    const float* b_out  = (const float*)d_in[6];
    const float* ln_g   = (const float*)d_in[7];
    const float* ln_b   = (const float*)d_in[8];
    float* out = (float*)d_out;

    char* ws = (char*)d_ws;
    size_t OFF_WINB  = 0;
    size_t OFF_QT    = OFF_WINB + (size_t)C * C * 2;          // 2 MiB
    size_t OFF_QPROJ = OFF_QT + (size_t)NPIX * C * 2;         // +32 MiB
    size_t OFF_KP    = OFF_QPROJ + (size_t)C * NPIX * 2;      // +32 MiB
    size_t OFF_VP    = OFF_KP + 4096;
    size_t OFF_KPROJ = OFF_VP + 4096;
    size_t OFF_VPROJ = OFF_KPROJ + 4096;
    size_t OFF_GK    = OFF_VPROJ + 4096;
    size_t OFF_GB    = OFF_GK + 4096;
    size_t OFF_WV    = OFF_GB + 256;
    size_t OFF_HMAX  = OFF_WV + 65536;
    size_t OFF_ATTN  = OFF_HMAX + 256;

    unsigned short* winb  = (unsigned short*)(ws + OFF_WINB);
    unsigned short* qt    = (unsigned short*)(ws + OFF_QT);
    unsigned short* qproj = (unsigned short*)(ws + OFF_QPROJ);
    float* kpmax = (float*)(ws + OFF_KP);
    float* vpmax = (float*)(ws + OFF_VP);
    float* kproj = (float*)(ws + OFF_KPROJ);
    float* vproj = (float*)(ws + OFF_VPROJ);
    float* gk    = (float*)(ws + OFF_GK);
    float* GB    = (float*)(ws + OFF_GB);
    float* wv    = (float*)(ws + OFF_WV);
    int*   hmax  = (int*)  (ws + OFF_HMAX);
    float* attn  = (float*)(ws + OFF_ATTN);

    k_init_hmax<<<1, 32, 0, stream>>>(hmax);
    k_maxpool<<<2048, 256, 0, stream>>>(keys, values, kpmax, vpmax);
    k_cvt_win<<<1024, 256, 0, stream>>>(w_in, winb);
    k_transpose_q<<<dim3(NPIX / 32, C / 32), 256, 0, stream>>>(query, qt);
    k_proj_kv<<<4, 256, 0, stream>>>(w_in, b_in, kpmax, vpmax, kproj, vproj);
    k_ln_k<<<1, 1024, 0, stream>>>(kproj, ln_g, ln_b, gk, GB);
    k_wv<<<4, 256, 0, stream>>>(w_out, vproj, wv);
    k_gemm_qproj<<<dim3(NPIX / 128, C / 256), 256, 0, stream>>>(winb, qt, qproj);
    k_attn<<<NPIX / 32, 256, 0, stream>>>(qproj, b_in, gk, GB, hmax, attn);
    k_out<<<dim3(NPIX / 256, C), 256, 0, stream>>>(wv, hmax, attn, b_out, out);

    (void)in_sizes; (void)n_in; (void)out_size; (void)ws_size;
}
